// Masked_Multi_Head_Attention_79912161509790
// MI455X (gfx1250) — matrix operand order repaired and checked by reading, not compiled, not run on hardware
//
#include <hip/hip_runtime.h>

typedef __bf16 bf16_t;
typedef __attribute__((ext_vector_type(16))) __bf16 v16bf;
typedef __attribute__((ext_vector_type(8)))  __bf16 v8bf;
typedef __attribute__((ext_vector_type(8)))  float  v8f;

#define BB   2
#define SEQ  2048
#define DM   1024
#define NH   16
#define HDIM 64

// ---------------------------------------------------------------- helpers

__device__ __forceinline__ v16bf cat16(v8bf lo, v8bf hi) {
  return __builtin_shufflevector(lo, hi, 0,1,2,3,4,5,6,7,8,9,10,11,12,13,14,15);
}

__device__ __forceinline__ v8f wmma_bf16(v16bf a, v16bf b, v8f c) {
  return __builtin_amdgcn_wmma_f32_16x16x32_bf16(false, a, false, b, (short)0, c,
                                                 false, false);
}

__device__ __forceinline__ float red_max16(float v) {
  v = fmaxf(v, __shfl_xor(v, 1, 32));
  v = fmaxf(v, __shfl_xor(v, 2, 32));
  v = fmaxf(v, __shfl_xor(v, 4, 32));
  v = fmaxf(v, __shfl_xor(v, 8, 32));
  return v;
}
__device__ __forceinline__ float red_sum16(float v) {
  v += __shfl_xor(v, 1, 32);
  v += __shfl_xor(v, 2, 32);
  v += __shfl_xor(v, 4, 32);
  v += __shfl_xor(v, 8, 32);
  return v;
}

// ---------------------------------------------------------------- fp32 -> bf16
__global__ __launch_bounds__(256)
void cvt_f32_bf16(const float* __restrict__ in, bf16_t* __restrict__ out, int n) {
  const int i = (blockIdx.x * 256 + threadIdx.x) * 8;
  if (i < n) {
    v8f v = *(const v8f*)(in + i);
    *(v8bf*)(out + i) = __builtin_convertvector(v, v8bf);
  }
}

// ---------------------------------------------------------------- GEMM (bf16 in)
// C[m][n] = sum_k A[m][k] * W[n][k] + bias[n]
// A: MxK bf16 row-major, W: NxK bf16 row-major, bias fp32.
// mode 0: fp32 out, row-major MxN            (output projection)
// mode 1: bf16 out, (B,H,S,HD) head-major    (Q, K)
// mode 2: bf16 out, (B,H,HD,S) transposed    (V)
// Block: 256 threads (8 waves as 2Mx4N). Block tile 128(M) x 256(N);
// wave tile 64x64 -> 16 WMMA per k-step, every fragment reused 4x.
__global__ __launch_bounds__(256)
void gemm_bf16_wmma(const bf16_t* __restrict__ A, const bf16_t* __restrict__ W,
                    const float* __restrict__ bias, int M, int N, int K,
                    float* __restrict__ outF, bf16_t* __restrict__ outB,
                    int mode)
{
  const int lane = threadIdx.x & 31;
  const int wave = threadIdx.x >> 5;
  const int wm   = wave & 1;            // M half (64 rows each)
  const int wn   = wave >> 1;           // N strip 0..3 (64 cols each)
  const int m0   = blockIdx.x * 128 + wm * 64;
  const int n0   = blockIdx.y * 256 + wn * 64;

  const int half  = lane >> 4;          // 0/1
  const int l16   = lane & 15;
  const int koff  = half * 8;           // unused; A-frag uses kbase too
  const int kbase = half * 16;          // B-frag k base

  v8f acc[4][4] = {};

  const bf16_t* Arow0 = A + (size_t)(m0 + l16) * K;

  for (int k0 = 0; k0 < K; k0 += 32) {
    __builtin_prefetch(Arow0 + k0 + 256, 0, 3);
    v16bf af[4];
#pragma unroll
    for (int i = 0; i < 4; ++i) {
      const bf16_t* Ar = Arow0 + (size_t)(i * 16) * K + k0;
      af[i] = cat16(*(const v8bf*)(Ar + kbase), *(const v8bf*)(Ar + 8 + kbase));
    }
#pragma unroll
    for (int t = 0; t < 4; ++t) {
      const bf16_t* Wrow = W + (size_t)(n0 + t * 16 + l16) * K + k0 + kbase;
      v16bf bf = *(const v16bf*)Wrow;      // 16 contiguous bf16 (32B)
#pragma unroll
      for (int i = 0; i < 4; ++i)
        acc[i][t] = wmma_bf16(af[i], bf, acc[i][t]);
    }
  }

#pragma unroll
  for (int i = 0; i < 4; ++i) {
#pragma unroll
    for (int t = 0; t < 4; ++t) {
      const int n  = n0 + t * 16 + l16;
      const float bv = bias[n];
#pragma unroll
      for (int r = 0; r < 8; ++r) {
        const int m = m0 + i * 16 + r + half * 8;
        const float val = acc[i][t][r] + bv;
        if (mode == 0) {
          outF[(size_t)m * N + n] = val;
        } else {
          const int b = m / SEQ, s = m % SEQ;
          const int h = n / HDIM, dd = n % HDIM;
          if (mode == 1)
            outB[(((size_t)b * NH + h) * SEQ + s) * HDIM + dd] = (bf16_t)val;
          else
            outB[(((size_t)b * NH + h) * HDIM + dd) * SEQ + s] = (bf16_t)val;
        }
      }
    }
  }
}

// ---------------------------------------------------------------- attention
// Q,K: (B,H,S,HD) bf16.  Vt: (B,H,HD,S) bf16.  kpm: (B,S) int.
// ctx out: (B,S,D) bf16.  One wave per 32-query tile (two 16-row subtiles
// sharing all K/V fragments); 4 waves / block.
__global__ __launch_bounds__(128)
void flash_attn_wmma(const bf16_t* __restrict__ Q, const bf16_t* __restrict__ Kc,
                     const bf16_t* __restrict__ Vt, const int* __restrict__ kpm,
                     bf16_t* __restrict__ ctx)
{
  __shared__ __align__(32) bf16_t pshared[4][2 * 16 * 32];

  const int lane  = threadIdx.x & 31;
  const int wave  = threadIdx.x >> 5;
  const int qtile = blockIdx.x * 4 + wave;
  const int h     = blockIdx.y;
  const int b     = blockIdx.z;
  const int q0    = qtile * 32;

  const bf16_t* Qh  = Q  + ((size_t)b * NH + h) * SEQ * HDIM;
  const bf16_t* Kh  = Kc + ((size_t)b * NH + h) * SEQ * HDIM;
  const bf16_t* Vh  = Vt + ((size_t)b * NH + h) * HDIM * SEQ;
  const int* kpmb   = kpm + b * SEQ;
  bf16_t* Pw        = pshared[wave];

  const int half  = lane >> 4;
  const int l16   = lane & 15;
  const int koff  = half * 8;
  const int kbase = half * 16;

  // Q A-fragments: [subtile u][d-chunk c]
  v16bf qf[2][2];
#pragma unroll
  for (int u = 0; u < 2; ++u) {
    const bf16_t* Qrow = Qh + (size_t)(q0 + u * 16 + l16) * HDIM;
#pragma unroll
    for (int c = 0; c < 2; ++c) {
      v8bf lo = *(const v8bf*)(Qrow + c * 32 + kbase);
      v8bf hi = *(const v8bf*)(Qrow + c * 32 + 8 + kbase);
      qf[u][c] = cat16(lo, hi);
    }
  }

  float mrun[2][8], lrun[2][8];
  v8f o[2][4] = {};
#pragma unroll
  for (int u = 0; u < 2; ++u)
#pragma unroll
    for (int r = 0; r < 8; ++r) { mrun[u][r] = -1e30f; lrun[u][r] = 0.f; }

  const float NEG = -__builtin_inff();
  const float SCL = 0.125f * 1.44269504f;         // 1/sqrt(64) * log2(e)
  const int kend = q0 + 32;                       // causal bound, multiple of 32

  for (int j0 = 0; j0 < kend; j0 += 32) {
    // ---- scores: two 16-key tiles x two query subtiles (k=64 as 2x32)
    v8f s[2][2] = {};                             // [u][key tile]
#pragma unroll
    for (int c = 0; c < 2; ++c) {
      v16bf kf0 = *(const v16bf*)(Kh + (size_t)(j0 + l16) * HDIM + c * 32 + kbase);
      s[0][0] = wmma_bf16(qf[0][c], kf0, s[0][0]);
      s[1][0] = wmma_bf16(qf[1][c], kf0, s[1][0]);
      v16bf kf1 = *(const v16bf*)(Kh + (size_t)(j0 + 16 + l16) * HDIM + c * 32 + kbase);
      s[0][1] = wmma_bf16(qf[0][c], kf1, s[0][1]);
      s[1][1] = wmma_bf16(qf[1][c], kf1, s[1][1]);
    }

    // ---- mask + online softmax (exp2 domain, scale folded in)
    const int key0 = j0 + l16;
    const int key1 = j0 + 16 + l16;
    const int pm0  = kpmb[key0];
    const int pm1  = kpmb[key1];

#pragma unroll
    for (int u = 0; u < 2; ++u) {
      bf16_t* P = Pw + u * (16 * 32);
#pragma unroll
      for (int r = 0; r < 8; ++r) {
        const int q = q0 + u * 16 + r + half * 8;
        float v0 = (key0 <= q && pm0 == 0) ? s[u][0][r] * SCL : NEG;
        float v1 = (key1 <= q && pm1 == 0) ? s[u][1][r] * SCL : NEG;
        const float mx   = red_max16(fmaxf(v0, v1));
        const float mnew = fmaxf(mrun[u][r], mx);   // finite (init -1e30)
        const float alpha = exp2f(mrun[u][r] - mnew);
        const float p0 = exp2f(v0 - mnew);
        const float p1 = exp2f(v1 - mnew);
        lrun[u][r] = lrun[u][r] * alpha + red_sum16(p0 + p1);
        mrun[u][r] = mnew;
#pragma unroll
        for (int t = 0; t < 4; ++t) o[u][t][r] *= alpha;
        const int prow = r + half * 8;
        P[prow * 32 + l16]      = (bf16_t)p0;
        P[prow * 32 + 16 + l16] = (bf16_t)p1;
      }
    }

    // ---- P: C-layout -> A-layout via LDS (same wave, dep via DScnt)
    v16bf pf[2];
#pragma unroll
    for (int u = 0; u < 2; ++u) {
      const bf16_t* Pr = Pw + u * (16 * 32) + l16 * 32;
      pf[u] = cat16(*(const v8bf*)(Pr + kbase), *(const v8bf*)(Pr + 8 + kbase));
    }

    // ---- O += P V  (V fragments shared by both query subtiles)
#pragma unroll
    for (int t = 0; t < 4; ++t) {
      v16bf vf = *(const v16bf*)(Vh + (size_t)(t * 16 + l16) * SEQ + j0 + kbase);
      o[0][t] = wmma_bf16(pf[0], vf, o[0][t]);
      o[1][t] = wmma_bf16(pf[1], vf, o[1][t]);
    }
  }

  // ---- normalize and emit context (B,S,D) bf16
#pragma unroll
  for (int u = 0; u < 2; ++u) {
#pragma unroll
    for (int r = 0; r < 8; ++r) {
      const float inv = 1.0f / lrun[u][r];
      const int q = q0 + u * 16 + r + half * 8;
      bf16_t* crow = ctx + ((size_t)b * SEQ + q) * DM + h * HDIM;
#pragma unroll
      for (int t = 0; t < 4; ++t)
        crow[t * 16 + l16] = (bf16_t)(o[u][t][r] * inv);
    }
  }
}

// ---------------------------------------------------------------- launch

extern "C" void kernel_launch(void* const* d_in, const int* in_sizes, int n_in,
                              void* d_out, int out_size, void* d_ws, size_t ws_size,
                              hipStream_t stream) {
  (void)in_sizes; (void)n_in; (void)out_size; (void)ws_size;
  const float* X   = (const float*)d_in[0];
  const int*   kpm = (const int*)d_in[1];
  const float* Wq  = (const float*)d_in[2];
  const float* bq  = (const float*)d_in[3];
  const float* Wk  = (const float*)d_in[4];
  const float* bk  = (const float*)d_in[5];
  const float* Wv  = (const float*)d_in[6];
  const float* bv  = (const float*)d_in[7];
  const float* Wo  = (const float*)d_in[8];
  const float* bo  = (const float*)d_in[9];

  const size_t actElems = (size_t)BB * SEQ * DM;          // 4,194,304
  const size_t wElems   = (size_t)DM * DM;                // 1,048,576
  bf16_t* Qw   = (bf16_t*)d_ws;
  bf16_t* Kw   = Qw + actElems;
  bf16_t* Vw   = Kw + actElems;
  bf16_t* ctxb = Vw + actElems;
  bf16_t* Xb   = ctxb + actElems;
  bf16_t* Wqb  = Xb + actElems;
  bf16_t* Wkb  = Wqb + wElems;
  bf16_t* Wvb  = Wkb + wElems;
  bf16_t* Wob  = Wvb + wElems;

  // one-shot fp32 -> bf16 conversion (serves all four GEMMs)
  cvt_f32_bf16<<<dim3(actElems / 8 / 256), 256, 0, stream>>>(X,  Xb,  (int)actElems);
  cvt_f32_bf16<<<dim3(wElems   / 8 / 256), 256, 0, stream>>>(Wq, Wqb, (int)wElems);
  cvt_f32_bf16<<<dim3(wElems   / 8 / 256), 256, 0, stream>>>(Wk, Wkb, (int)wElems);
  cvt_f32_bf16<<<dim3(wElems   / 8 / 256), 256, 0, stream>>>(Wv, Wvb, (int)wElems);
  cvt_f32_bf16<<<dim3(wElems   / 8 / 256), 256, 0, stream>>>(Wo, Wob, (int)wElems);

  const int M = BB * SEQ;                                 // 4096
  const int N = DM, K = DM;                               // 1024, 1024
  dim3 gblk(256);
  dim3 ggrd(M / 128, N / 256);                            // (32, 4)

  gemm_bf16_wmma<<<ggrd, gblk, 0, stream>>>(Xb, Wqb, bq, M, N, K, nullptr, Qw, 1);
  gemm_bf16_wmma<<<ggrd, gblk, 0, stream>>>(Xb, Wkb, bk, M, N, K, nullptr, Kw, 1);
  gemm_bf16_wmma<<<ggrd, gblk, 0, stream>>>(Xb, Wvb, bv, M, N, K, nullptr, Vw, 2);

  flash_attn_wmma<<<dim3(SEQ / 32 / 4, NH, BB), 128, 0, stream>>>(Qw, Kw, Vw, kpm, ctxb);

  gemm_bf16_wmma<<<ggrd, gblk, 0, stream>>>(ctxb, Wob, bo, M, N, K, (float*)d_out, nullptr, 0);
}
